// GraphConv_31439160607326
// MI455X (gfx1250) — compile-verified, matrix operand order checked by reading, not run on hardware
//
#include <hip/hip_runtime.h>
#include <hip/hip_bf16.h>
#include <float.h>

typedef __attribute__((ext_vector_type(16))) __bf16 v16bf;
typedef __attribute__((ext_vector_type(8)))  float  v8f;

#define N_NODES  50000
#define N_EDGES  800000
#define N_GRAPHS 256

static inline int cdiv(long long a, long long b) { return (int)((a + b - 1) / b); }

// ---------------- utility kernels ----------------

__global__ void fill_kernel(float* p, float v, long long n) {
    long long i = (long long)blockIdx.x * blockDim.x + threadIdx.x;
    if (i < n) p[i] = v;
}

__global__ void neg_copy_kernel(const float* __restrict__ src, float* __restrict__ dst, long long n) {
    long long i = (long long)blockIdx.x * blockDim.x + threadIdx.x;
    if (i < n) dst[i] = -src[i];
}

// ---------------- graph normalization ----------------

__global__ void deg_kernel(const long long* __restrict__ row, float* __restrict__ deg) {
    int e = blockIdx.x * blockDim.x + threadIdx.x;
    if (e < N_EDGES) atomicAdd(&deg[(int)row[e]], 1.0f);
}

__global__ void dinv_kernel(float* __restrict__ deg) {
    int i = blockIdx.x * blockDim.x + threadIdx.x;
    if (i < N_NODES) {
        float d = deg[i];
        deg[i] = (d > 0.0f) ? rsqrtf(fmaxf(d, 1e-12f)) : 0.0f;
    }
}

__global__ void norm_kernel(const long long* __restrict__ row, const long long* __restrict__ col,
                            const float* __restrict__ dinv, float* __restrict__ nrm) {
    int e = blockIdx.x * blockDim.x + threadIdx.x;
    if (e < N_EDGES) nrm[e] = -(dinv[(int)row[e]] * dinv[(int)col[e]]);
}

// out[col[e]*F + 4c..] += scale * norm[e] * t[row[e]*F + 4c..]   (one thread = edge x 4 feats)
__global__ void prop_kernel(const long long* __restrict__ row, const long long* __restrict__ col,
                            const float* __restrict__ nrm, const float* __restrict__ t,
                            float* __restrict__ out, int F4, float scale) {
    long long idx = (long long)blockIdx.x * blockDim.x + threadIdx.x;
    long long total = (long long)N_EDGES * F4;
    if (idx >= total) return;
    int e = (int)(idx / F4);
    int c = (int)(idx - (long long)e * F4);
    float w = scale * nrm[e];
    const float4* tp = (const float4*)(t + ((long long)row[e] * F4 + c) * 4);
    float4 v = *tp;
    float* op = out + ((long long)col[e] * F4 + c) * 4;
    atomicAdd(op + 0, w * v.x);
    atomicAdd(op + 1, w * v.y);
    atomicAdd(op + 2, w * v.z);
    atomicAdd(op + 3, w * v.w);
}

// ---------------- weight packing: f32 [nTerms][K][Nout] -> bf16 B-fragments ----------------
// fragIdx = ((s*NT + nt)*KB + kb)*32 + lane ; each fragment holds 16 bf16 in WMMA B lane order.
__global__ void pack_w_kernel(const float* __restrict__ W, __bf16* __restrict__ P,
                              int K, int Nout, int nTerms) {
    int KB = K >> 5, NT = Nout >> 4;
    int total = nTerms * NT * KB * 32;
    int idx = blockIdx.x * blockDim.x + threadIdx.x;
    if (idx >= total) return;
    int lane = idx & 31;
    int r = idx >> 5;
    int kb = r % KB; r /= KB;
    int nt = r % NT;
    int s  = r / NT;
    int half = lane >> 4, l15 = lane & 15;
    const float* Wp = W + (size_t)s * K * Nout + nt * 16 + l15;
    v16bf v;
#pragma unroll
    for (int j = 0; j < 8; ++j) {
        int k0 = kb * 32 + 8 * half + j;
        int k1 = kb * 32 + 16 + 8 * half + j;
        v[j]     = (__bf16)Wp[(size_t)k0 * Nout];
        v[j + 8] = (__bf16)Wp[(size_t)k1 * Nout];
    }
    *(v16bf*)(P + (size_t)idx * 16) = v;
}

// ---------------- WMMA GEMM ----------------
// out = sum_s A_s[M x K] @ W[s] + bias, activation v = max(v,0) + negSlope*min(v,0).
// Block = 128 threads = 4 waves; wave w handles mTile = blockIdx.x*4+w and
// 4 consecutive N-tiles (blockIdx.y*4 ..+3), reusing each A fragment across 4 WMMAs.
__global__ void wmma_gemm_kernel(const float* __restrict__ A0, const float* __restrict__ A1,
                                 const float* __restrict__ A2, const __bf16* __restrict__ P,
                                 const float* __restrict__ bias, float* __restrict__ out,
                                 int K, int Nout, int nTerms, float negSlope, int Mtiles) {
    const int lane = threadIdx.x & 31;
    const int wave = threadIdx.x >> 5;
    const int mTile = blockIdx.x * 4 + wave;
    if (mTile >= Mtiles) return;                    // wave-uniform
    const int half = lane >> 4, l15 = lane & 15;
    const int mBase = mTile * 16;
    const int nTile0 = blockIdx.y * 4;
    const int KB = K >> 5, NT = Nout >> 4;
    const v16bf* Bf = (const v16bf*)P;

    v8f acc0 = {}, acc1 = {}, acc2 = {}, acc3 = {};
    for (int s = 0; s < nTerms; ++s) {
        const float* A = (s == 0) ? A0 : (s == 1) ? A1 : A2;
        const float* arowBase = A + (size_t)(mBase + l15) * K + 8 * half;
        for (int kb = 0; kb < KB; ++kb) {
            const float4* ap = (const float4*)(arowBase + kb * 32);
            float4 f0 = ap[0], f1 = ap[1], f2 = ap[4], f3 = ap[5];
            v16bf af;
            af[0]  = (__bf16)f0.x; af[1]  = (__bf16)f0.y; af[2]  = (__bf16)f0.z; af[3]  = (__bf16)f0.w;
            af[4]  = (__bf16)f1.x; af[5]  = (__bf16)f1.y; af[6]  = (__bf16)f1.z; af[7]  = (__bf16)f1.w;
            af[8]  = (__bf16)f2.x; af[9]  = (__bf16)f2.y; af[10] = (__bf16)f2.z; af[11] = (__bf16)f2.w;
            af[12] = (__bf16)f3.x; af[13] = (__bf16)f3.y; af[14] = (__bf16)f3.z; af[15] = (__bf16)f3.w;

            size_t fb = (((size_t)s * NT + nTile0) * KB + kb) * 32 + lane;
            size_t step = (size_t)KB * 32;          // stride between consecutive N-tiles
            v16bf b0 = Bf[fb];
            acc0 = __builtin_amdgcn_wmma_f32_16x16x32_bf16(false, af, false, b0, (short)0, acc0, false, false);
            v16bf b1 = Bf[fb + step];
            acc1 = __builtin_amdgcn_wmma_f32_16x16x32_bf16(false, af, false, b1, (short)0, acc1, false, false);
            v16bf b2 = Bf[fb + 2 * step];
            acc2 = __builtin_amdgcn_wmma_f32_16x16x32_bf16(false, af, false, b2, (short)0, acc2, false, false);
            v16bf b3 = Bf[fb + 3 * step];
            acc3 = __builtin_amdgcn_wmma_f32_16x16x32_bf16(false, af, false, b3, (short)0, acc3, false, false);
        }
    }

#pragma unroll
    for (int q = 0; q < 4; ++q) {
        v8f acc = (q == 0) ? acc0 : (q == 1) ? acc1 : (q == 2) ? acc2 : acc3;
        int nc = (nTile0 + q) * 16 + l15;
        float bv = bias[nc];
#pragma unroll
        for (int r = 0; r < 8; ++r) {
            float v = acc[r] + bv;
            v = fmaxf(v, 0.0f) + negSlope * fminf(v, 0.0f);
            out[(size_t)(mBase + r + 8 * half) * Nout + nc] = v;
        }
    }
}

// ---------------- global max pool ----------------

__device__ inline void atomicMaxFloat(float* addr, float val) {
    if (val >= 0.0f) atomicMax((int*)addr, __float_as_int(val));
    else             atomicMin((unsigned int*)addr, __float_as_uint(val));
}

__global__ void pool_max_kernel(const float* __restrict__ h, const long long* __restrict__ batch,
                                float* __restrict__ g) {
    long long idx = (long long)blockIdx.x * blockDim.x + threadIdx.x;   // N_NODES * 64
    long long total = (long long)N_NODES * 64;
    if (idx >= total) return;
    int n = (int)(idx >> 6);
    int c = (int)(idx & 63);
    float4 v = *(const float4*)(h + ((long long)n * 64 + c) * 4);
    float* gp = g + ((long long)batch[n] * 64 + c) * 4;
    atomicMaxFloat(gp + 0, v.x);
    atomicMaxFloat(gp + 1, v.y);
    atomicMaxFloat(gp + 2, v.z);
    atomicMaxFloat(gp + 3, v.w);
}

// ---------------- final tiny FC (512 -> 4) ----------------

__global__ void fc3_kernel(const float* __restrict__ x, const float* __restrict__ W,
                           const float* __restrict__ b, float* __restrict__ out) {
    int idx = blockIdx.x * blockDim.x + threadIdx.x;   // 256 * 4
    if (idx >= N_GRAPHS * 4) return;
    int g = idx >> 2, c = idx & 3;
    float acc = b[c];
    const float* xr = x + (size_t)g * 512;
    for (int k = 0; k < 512; ++k) acc += xr[k] * W[k * 4 + c];
    out[idx] = acc;
}

// ---------------- orchestration ----------------

extern "C" void kernel_launch(void* const* d_in, const int* in_sizes, int n_in,
                              void* d_out, int out_size, void* d_ws, size_t ws_size,
                              hipStream_t stream) {
    const float*     x     = (const float*)d_in[0];
    const long long* ei    = (const long long*)d_in[1];    // [2, E]
    const long long* row   = ei;
    const long long* col   = ei + N_EDGES;
    const long long* batch = (const long long*)d_in[2];
    const float* W1 = (const float*)d_in[3];  const float* b1 = (const float*)d_in[4];
    const float* W2 = (const float*)d_in[5];  const float* b2 = (const float*)d_in[6];
    const float* W3 = (const float*)d_in[7];  const float* b3 = (const float*)d_in[8];
    const float* Wc1 = (const float*)d_in[9];  const float* bc1 = (const float*)d_in[10];
    const float* Wc2 = (const float*)d_in[11]; const float* bc2 = (const float*)d_in[12];
    const float* Wc3 = (const float*)d_in[13]; const float* bc3 = (const float*)d_in[14];
    float* out = (float*)d_out;

    // workspace layout (floats, all segment sizes multiples of 8 -> 32B alignment kept)
    float* ws  = (float*)d_ws;
    float* deg = ws;                                   // 50000 (becomes dinv in place)
    float* nrm = deg + N_NODES;                        // 800000
    float* t1  = nrm + N_EDGES;                        // 50000*128 max
    float* t2  = t1 + (size_t)N_NODES * 128;           // 50000*128 max
    float* h1  = t2 + (size_t)N_NODES * 128;           // 50000*64
    float* h2  = h1 + (size_t)N_NODES * 64;            // 50000*128
    float* h3  = h2 + (size_t)N_NODES * 128;           // 50000*256
    float* g   = h3 + (size_t)N_NODES * 256;           // 256*256
    float* m1  = g  + (size_t)N_GRAPHS * 256;          // 256*1024
    float* m2  = m1 + (size_t)N_GRAPHS * 1024;         // 256*512
    __bf16* pw = (__bf16*)(m2 + (size_t)N_GRAPHS * 512);  // max 1024*512 bf16 (reused per GEMM)

    const int B = 256;

    // normalization
    fill_kernel<<<cdiv(N_NODES, B), B, 0, stream>>>(deg, 0.0f, N_NODES);
    deg_kernel<<<cdiv(N_EDGES, B), B, 0, stream>>>(row, deg);
    dinv_kernel<<<cdiv(N_NODES, B), B, 0, stream>>>(deg);
    norm_kernel<<<cdiv(N_EDGES, B), B, 0, stream>>>(row, col, deg, nrm);

    // pack weights + GEMM launcher
    auto gemm = [&](const float* A0, const float* A1, const float* A2, const float* W,
                    const float* bias, float* h, int M, int K, int Nout, int nTerms,
                    float negSlope) {
        int packN = nTerms * (Nout / 16) * (K / 32) * 32;
        pack_w_kernel<<<cdiv(packN, B), B, 0, stream>>>(W, pw, K, Nout, nTerms);
        int Mtiles = M / 16;
        dim3 grid(cdiv(Mtiles, 4), Nout / 64);
        wmma_gemm_kernel<<<grid, 128, 0, stream>>>(A0, A1, A2, pw, bias, h,
                                                   K, Nout, nTerms, negSlope, Mtiles);
    };

    // one ChebConv layer: t0 = input, t1 = prop(t0), t2 = 2*prop(t1) - t0, fused GEMM
    auto cheb_layer = [&](const float* t0, const float* W, const float* bias,
                          float* h, int K, int Nout, float negSlope) {
        long long nf = (long long)N_NODES * K;
        fill_kernel<<<cdiv(nf, B), B, 0, stream>>>(t1, 0.0f, nf);
        prop_kernel<<<cdiv((long long)N_EDGES * (K / 4), B), B, 0, stream>>>(row, col, nrm, t0, t1, K / 4, 1.0f);
        neg_copy_kernel<<<cdiv(nf, B), B, 0, stream>>>(t0, t2, nf);
        prop_kernel<<<cdiv((long long)N_EDGES * (K / 4), B), B, 0, stream>>>(row, col, nrm, t1, t2, K / 4, 2.0f);
        gemm(t0, t1, t2, W, bias, h, N_NODES, K, Nout, 3, negSlope);
    };

    cheb_layer(x,  W1, b1, h1, 128, 64,  0.01f);   // LeakyReLU
    cheb_layer(h1, W2, b2, h2, 64,  128, 0.01f);   // LeakyReLU
    cheb_layer(h2, W3, b3, h3, 128, 256, 1.0f);    // linear

    // global max pool per graph
    fill_kernel<<<cdiv((long long)N_GRAPHS * 256, B), B, 0, stream>>>(g, -FLT_MAX, (long long)N_GRAPHS * 256);
    pool_max_kernel<<<cdiv((long long)N_NODES * 64, B), B, 0, stream>>>(h3, batch, g);

    // MLP head
    gemm(g,  g,  g,  Wc1, bc1, m1, N_GRAPHS, 256,  1024, 1, 0.0f);  // ReLU
    gemm(m1, m1, m1, Wc2, bc2, m2, N_GRAPHS, 1024, 512,  1, 0.0f);  // ReLU
    fc3_kernel<<<cdiv(N_GRAPHS * 4, B), B, 0, stream>>>(m2, Wc3, bc3, out);
}
